// RKNLayer_31233002177247
// MI455X (gfx1250) — compile-verified, matrix operand order checked by reading, not run on hardware
//
#include <hip/hip_runtime.h>
#include <hip/hip_bf16.h>
#include <cmath>

typedef __bf16 bf16;
typedef __attribute__((ext_vector_type(16))) bf16  v16bf;
typedef __attribute__((ext_vector_type(8)))  bf16  v8bf;
typedef __attribute__((ext_vector_type(2)))  bf16  v2bf;
typedef __attribute__((ext_vector_type(8)))  float v8f;
typedef __attribute__((ext_vector_type(4)))  float v4f;
typedef __attribute__((ext_vector_type(2)))  float v2f;

#define B_   256
#define T_   256
#define M_   128
#define K_   32
#define U_   64
#define N_   256
#define TILE_B 16
#define NBLK   (B_ / TILE_B)     // 16 workgroups
#define NTHREADS 512             // 16 wave32
#define APITCH  4104             // A-band LDS pitch (bf16): 8208B = 4 dwords mod 64 banks
#define Z16PITCH 264             // z bf16 pitch: 528B = 4 dwords mod 64 banks
#define GPITCH  196              // gate f32 pitch: 784B = 4 dwords mod 64 banks
#define ABFPITCH 40              // alpha bf16 pitch: 80B = 20 dwords mod 64 banks
#define DKTPITCH 68              // dkT f32 pitch: 272B = 4 dwords mod 64 banks

// ---------------- LDS layout (bytes) ----------------
constexpr unsigned OFF_Z16  = 0;                              // 2 x 16 x Z16PITCH bf16 (ping-pong)
constexpr unsigned OFF_X    = OFF_Z16  + 2 * 16 * Z16PITCH * 2;
constexpr unsigned OFF_G    = OFF_X    + 4096 * 4;            // 16 x GPITCH f32 (i|c|o)
constexpr unsigned OFF_H    = OFF_G    + 16 * GPITCH * 4;     // 16x64 f32
constexpr unsigned OFF_ABF  = OFF_H    + 1024 * 4;            // 16 x ABFPITCH bf16 alpha
constexpr unsigned OFF_AB   = OFF_ABF  + 16 * ABFPITCH * 2;   // 16 x APITCH bf16 A-band
constexpr unsigned OFF_SU   = OFF_AB   + 16 * APITCH * 2;
constexpr unsigned OFF_SL   = OFF_SU   + 2048 * 4;
constexpr unsigned OFF_SS   = OFF_SL   + 2048 * 4;
constexpr unsigned OFF_STU  = OFF_SS   + 2048 * 4;
constexpr unsigned OFF_STL  = OFF_STU  + 2048 * 4;
constexpr unsigned OFF_DKT  = OFF_STL  + 2048 * 4;            // 32 x DKTPITCH f32
constexpr unsigned OFF_LB   = OFF_DKT  + 32 * DKTPITCH * 4;   // 256 f32 lstm bias
constexpr unsigned OFF_DB   = OFF_LB   + 256 * 4;             // 32 f32 dense bias
constexpr unsigned LDS_TOTAL= OFF_DB   + 32 * 4;              // ~228 KB (< 320 KB/WGP)

// ---------------- fast math helpers ----------------
__device__ __forceinline__ float fast_rcp(float x) {
#if __has_builtin(__builtin_amdgcn_rcpf)
  return __builtin_amdgcn_rcpf(x);
#else
  return 1.f / x;
#endif
}
__device__ __forceinline__ float fast_tanh(float x) {
#if __has_builtin(__builtin_amdgcn_tanhf)
  return __builtin_amdgcn_tanhf(x);
#else
  return tanhf(x);
#endif
}
__device__ __forceinline__ float sigm(float x) { return fast_rcp(1.f + __expf(-x)); }

// ---------------- fragment helpers ----------------
// Gather a 16xK tile of a row-major bf16 LDS matrix into a fragment, per the ISA
// 16-bit striped layout: lanes 0-15 (row=lane) hold K {0..7,16..23}; lanes 16-31
// hold K {8..15,24..31}.  Used for both operands (same lane-major convention).
__device__ __forceinline__ v16bf frag_lds(const bf16* base, int pitch, int kofs, int lane) {
  int row = lane & 15;
  const bf16* p = base + row * pitch + kofs + ((lane >= 16) ? 8 : 0);
  v8bf lo = *(const v8bf*)p;          // 16B ds load
  v8bf hi = *(const v8bf*)(p + 16);   // 16B ds load
  v16bf a;
#pragma unroll
  for (int e = 0; e < 8; ++e) { a[e] = lo[e]; a[e + 8] = hi[e]; }
  return a;
}

// Pre-swizzled fragment in global memory: 512 bf16 per 16x32 tile, lane-major.
__device__ __forceinline__ v16bf frag_glb(const bf16* __restrict__ g, int tile, int lane) {
  return *(const v16bf*)(g + (size_t)tile * 512 + (size_t)lane * 16);
}

// Pack a f32 C-fragment row-chunk to bf16 and store as one ds_store_b128.
__device__ __forceinline__ void store_pack(bf16* dst, v8f a) {
  v8bf p;
#pragma unroll
  for (int j = 0; j < 8; ++j) p[j] = (bf16)a[j];
  *(v8bf*)dst = p;
}

// ---------------- prep: swizzle lstm_kernel (256x256) -> bf16 fragment tiles ----------------
__global__ void prep_w(const float* __restrict__ Wk, bf16* __restrict__ Wsw) {
  int tid = blockIdx.x * blockDim.x + threadIdx.x;   // 65536 elements
  int tile = tid >> 9, within = tid & 511;
  int lane = within >> 4, idx = within & 15;
  int ntile = tile >> 3, kt = tile & 7;
  int nl = lane & 15;
  int kl = (idx & 7) + ((idx >> 3) << 4) + ((lane >= 16) ? 8 : 0);
  int k = kt * 32 + kl, n = ntile * 16 + nl;
  Wsw[tid] = (bf16)Wk[k * 256 + n];
}

// ---------------- prep: banded, masked B matrices -> bf16 fragment tiles ----------------
__global__ void prep_bb(const float* __restrict__ B11, const float* __restrict__ B12,
                        const float* __restrict__ B21, const float* __restrict__ B22,
                        const int* __restrict__ band, bf16* __restrict__ out) {
  int tid = blockIdx.x * blockDim.x + threadIdx.x;   // 131072 elements
  int tile = tid >> 9, within = tid & 511;           // tile == ntile (single k-tile, K=32)
  int lane = within >> 4, idx = within & 15;
  int k = (idx & 7) + ((idx >> 3) << 4) + ((lane >= 16) ? 8 : 0);
  int c = tile * 16 + (lane & 15);
  int mat = c >> 10, i = (c >> 3) & 127, d = c & 7;
  float v = 0.f;
  if (d < 7) {
    int j = i + d - 3;
    int ad = d - 3; if (ad < 0) ad = -ad;
    if (j >= 0 && j < 128 && ad <= *band) {
      const float* Bm = (mat == 0) ? B11 : (mat == 1) ? B12 : (mat == 2) ? B21 : B22;
      v = Bm[k * (M_ * M_) + i * M_ + j];
    }
  }
  out[tid] = (bf16)v;
}

// ---------------- persistent scan kernel: one WG per 16-batch tile ----------------
__global__ __launch_bounds__(NTHREADS)
void rkn_scan(const float* __restrict__ x, const float* __restrict__ logu,
              const float* __restrict__ logl, const float* __restrict__ lbias_g,
              const float* __restrict__ dk_g, const float* __restrict__ dbias_g,
              const bf16* __restrict__ Wsw, const bf16* __restrict__ Bbsw,
              float* __restrict__ out) {
  extern __shared__ char smem[];
  bf16*  z16  = (bf16*) (smem + OFF_Z16);
  float* xbuf = (float*)(smem + OFF_X);
  float* gbuf = (float*)(smem + OFF_G);
  float* hbuf = (float*)(smem + OFF_H);
  bf16*  abf  = (bf16*) (smem + OFF_ABF);
  bf16*  abnd = (bf16*) (smem + OFF_AB);
  float* su   = (float*)(smem + OFF_SU);
  float* sl   = (float*)(smem + OFF_SL);
  float* ss   = (float*)(smem + OFF_SS);
  float* stu  = (float*)(smem + OFF_STU);
  float* stl  = (float*)(smem + OFF_STL);
  float* dkt  = (float*)(smem + OFF_DKT);
  float* lb   = (float*)(smem + OFF_LB);
  float* db   = (float*)(smem + OFF_DB);

  const int tid = threadIdx.x, w = tid >> 5, lane = tid & 31;
  const int bblk = blockIdx.x;

  // ---- init recurrent state + cache small weights ----
  for (int e = tid; e < 2 * 16 * Z16PITCH; e += NTHREADS) z16[e] = (bf16)0.f;
  for (int e = tid; e < 2048; e += NTHREADS) {
    su[e] = 10.f; sl[e] = 10.f; ss[e] = 0.f;
    int b = e >> 7, i = e & 127, bg = bblk * TILE_B + b;
    stu[e] = __expf(logu[bg * M_ + i]);
    stl[e] = __expf(logl[bg * M_ + i]);
  }
  for (int e = tid; e < 2048; e += NTHREADS) {          // dkT[k][u] = dk[u][k]
    int k = e & 31, u = e >> 5;
    dkt[k * DKTPITCH + u] = dk_g[u * 32 + k];
  }
  if (tid < 256) lb[tid] = lbias_g[tid];
  if (tid < 32)  db[tid] = dbias_g[tid];

  // ---- preload t-invariant LSTM W fragments into registers (waves 0..11) ----
  const int r = w >> 2, ti = w & 3;                     // r: 0=i,1=c,2=o gate region
  const int nt = ((r == 0) ? 0 : (r == 1) ? 8 : 12) + ti;
  v16bf wa[8];
  if (w < 12) {
#pragma unroll
    for (int kt = 0; kt < 8; ++kt) wa[kt] = frag_glb(Wsw, nt * 8 + kt, lane);
  }
  __syncthreads();

  for (int t = 0; t < T_; ++t) {
    const bf16* zc = z16 + (t & 1) * (16 * Z16PITCH);
    bf16*       zn = z16 + ((t + 1) & 1) * (16 * Z16PITCH);

    // (2) LSTM GEMM  D = W^T(192x256) * z^T(256x16) on waves 0..11 (W in VGPRs);
    //     waves 12..15 stage x_t with one batched load burst.
    if (w < 12) {
      v8f acc = {};
#pragma unroll
      for (int kt = 0; kt < 8; ++kt) {
        v16bf zb = frag_lds(zc, Z16PITCH, kt * 32, lane);
        acc = __builtin_amdgcn_wmma_f32_16x16x32_bf16(false, wa[kt], false, zb,
                                                      (short)0, acc, false, false);
      }
      int b = lane & 15, r0 = (lane >= 16) ? 8 : 0;
      float* gd = gbuf + b * GPITCH + r * 64 + ti * 16 + r0;
      v4f lo = {acc[0], acc[1], acc[2], acc[3]};
      v4f hi = {acc[4], acc[5], acc[6], acc[7]};
      *(v4f*)gd = lo; *(v4f*)(gd + 4) = hi;
    } else {
      int tl = tid - 384;
      v4f tmp[8];
#pragma unroll
      for (int it = 0; it < 8; ++it) {
        int e = tl + it * 128, b = e >> 6, c4 = (e & 63) << 2;
        tmp[it] = *(const v4f*)(x + ((size_t)(bblk * TILE_B + b) * T_ + t) * N_ + c4);
      }
#pragma unroll
      for (int it = 0; it < 8; ++it) {
        int e = tl + it * 128, b = e >> 6, c4 = (e & 63) << 2;
        *(v4f*)(xbuf + b * 256 + c4) = tmp[it];
      }
    }
    __syncthreads();

    // (3) gates + h
#pragma unroll
    for (int q = 0; q < 2; ++q) {
      int e = tid * 2 + q, b = e >> 6, u = e & 63;
      const float* gb = gbuf + b * GPITCH;
      float ig = sigm(gb[u]       + lb[u]);
      float cg = fast_tanh(gb[64 + u] + lb[128 + u]);
      float og = sigm(gb[128 + u] + lb[192 + u]);
      hbuf[b * 64 + u] = og * fast_tanh(ig * cg);
    }
    __syncthreads();

    // (4) alpha logits + softmax: wave w owns batch row w, lane = k
    {
      const float* dr = dkt + lane * DKTPITCH;
      const float* hb = hbuf + w * 64;
      float acc = db[lane];
#pragma unroll
      for (int u4 = 0; u4 < 16; ++u4) {
        v4f hv = *(const v4f*)(hb + u4 * 4);
        v4f dv = *(const v4f*)(dr + u4 * 4);
        acc = fmaf(hv[0], dv[0], acc); acc = fmaf(hv[1], dv[1], acc);
        acc = fmaf(hv[2], dv[2], acc); acc = fmaf(hv[3], dv[3], acc);
      }
      float m = acc;
#pragma unroll
      for (int off = 16; off > 0; off >>= 1) m = fmaxf(m, __shfl_xor(m, off, 32));
      float ev = __expf(acc - m);
      float s = ev;
#pragma unroll
      for (int off = 16; off > 0; off >>= 1) s += __shfl_xor(s, off, 32);
      abf[w * ABFPITCH + lane] = (bf16)(ev * fast_rcp(s));
    }
    __syncthreads();

    // (5) A-band GEMM  D = Bband^T(4096x32) * alpha^T(32x16): 16 tiles/wave.
    //     The wave's base TILE INDEX is laundered through an opaque asm barrier
    //     inside the t-loop: LICM cannot hoist the fragment loads (which would
    //     make them long-lived and spill), yet the pointer keeps its global
    //     address space so these stay global_load_b128 tracked by LOADcnt only.
    //     Each half is one clause-batched L2 burst, then 8 wmma with a 1-stage
    //     software pipeline so tile tt's pack fills tile tt+1's hazard window.
    {
      int btile = w * 16;
      asm volatile("" : "+v"(btile));                  // opaque: defeat LICM
      v16bf aal = frag_lds(abf, ABFPITCH, 0, lane);
      int b = lane & 15, r0 = (lane >= 16) ? 8 : 0;
      bf16* arow = abnd + b * APITCH + r0 + w * 256;   // 16 tiles * 16 cols
#pragma unroll
      for (int half = 0; half < 2; ++half) {
        v16bf bm[8];
#pragma unroll
        for (int tt = 0; tt < 8; ++tt)
          bm[tt] = frag_glb(Bbsw, btile + half * 8 + tt, lane);
        v8f prev = __builtin_amdgcn_wmma_f32_16x16x32_bf16(false, bm[0], false, aal,
                                                           (short)0, v8f{}, false, false);
#pragma unroll
        for (int tt = 1; tt < 8; ++tt) {
          v8f cur = __builtin_amdgcn_wmma_f32_16x16x32_bf16(false, bm[tt], false, aal,
                                                            (short)0, v8f{}, false, false);
          store_pack(arow + (half * 8 + tt - 1) * 16, prev);
          prev = cur;
        }
        store_pack(arow + (half * 8 + 7) * 16, prev);
      }
    }
    __syncthreads();

    // (6) banded predict + posterior: each thread owns (b, 2i, 2i+1); band rows
    //     loaded per sub-row (4 v8bf at a time) to keep pressure low, z taps
    //     shared between the two rows, state and outputs as 2-wide vectors.
#pragma unroll
    for (int q = 0; q < 2; ++q) {
      int dp = tid + q * NTHREADS;                     // 0..1023
      int b = dp >> 6, i0 = (dp & 63) << 1;
      const bf16* ab = abnd + b * APITCH + i0 * 8;
      const bf16* zbp = zc + b * Z16PITCH;
      float zu[8], zl[8];
#pragma unroll
      for (int d = 0; d < 8; ++d) {
        int j = i0 + d - 3;
        int jc = (j < 0) ? 0 : (j > 127 ? 127 : j);    // A is 0 out of range anyway
        zu[d] = (float)zbp[jc];
        zl[d] = (float)zbp[128 + jc];
      }
      int e = b * 128 + i0;
      v2f suv = *(const v2f*)(su + e), slv = *(const v2f*)(sl + e);
      v2f ssv = *(const v2f*)(ss + e);
      v2f stuv = *(const v2f*)(stu + e), stlv = *(const v2f*)(stl + e);
      v2f amv = *(const v2f*)(xbuf + b * 256 + i0);
      v2f asv = *(const v2f*)(xbuf + b * 256 + 128 + i0);
      v2f sunew, slnew, ssnew, ou, ol;
#pragma unroll
      for (int sub = 0; sub < 2; ++sub) {
        v8bf m11 = *(const v8bf*)(ab + sub * 8);
        v8bf m12 = *(const v8bf*)(ab + 1024 + sub * 8);
        v8bf m21 = *(const v8bf*)(ab + 2048 + sub * 8);
        v8bf m22 = *(const v8bf*)(ab + 3072 + sub * 8);
        float zpu = 0, zpl = 0, s11 = 0, s1112 = 0, s12 = 0, s21 = 0, s2221 = 0,
              s22 = 0, s2111 = 0, sxv = 0, s2212 = 0;
#pragma unroll
        for (int d = 0; d < 7; ++d) {
          float a11 = (float)m11[d], a12 = (float)m12[d];
          float a21 = (float)m21[d], a22 = (float)m22[d];
          float zud = zu[d + sub], zld = zl[d + sub];
          zpu += a11 * zud + a12 * zld;
          zpl += a21 * zud + a22 * zld;
          s11 += a11 * a11; s1112 += a11 * a12; s12 += a12 * a12;
          s21 += a21 * a21; s2221 += a22 * a21; s22 += a22 * a22;
          s2111 += a21 * a11; sxv += a22 * a11 + a21 * a12; s2212 += a22 * a12;
        }
        float su_ = suv[sub], sl_ = slv[sub], ss_ = ssv[sub];
        float su_p = s11 * su_ + 2.f * s1112 * ss_ + s12 * sl_ + stuv[sub];
        float sl_p = s21 * su_ + 2.f * s2221 * ss_ + s22 * sl_ + stlv[sub];
        float ss_p = s2111 * su_ + sxv * ss_ + s2212 * sl_;
        float denom = su_p + asv[sub];
        float inv = fast_rcp(denom);
        float qu = su_p * inv, ql = ss_p * inv;
        float res = amv[sub] - zpu;
        ou[sub] = zpu + qu * res;
        ol[sub] = zpl + ql * res;
        sunew[sub] = (1.f - qu) * su_p;
        slnew[sub] = sl_p - qu * ss_p;
        ssnew[sub] = (1.f - qu) * ss_p;
      }
      *(v2f*)(su + e) = sunew; *(v2f*)(sl + e) = slnew; *(v2f*)(ss + e) = ssnew;
      v2bf zu16 = {(bf16)ou[0], (bf16)ou[1]};
      v2bf zl16 = {(bf16)ol[0], (bf16)ol[1]};
      *(v2bf*)(zn + b * Z16PITCH + i0) = zu16;
      *(v2bf*)(zn + b * Z16PITCH + 128 + i0) = zl16;
      float* og = out + ((size_t)(bblk * TILE_B + b) * T_ + t) * N_;
      *(v2f*)(og + i0) = ou;
      *(v2f*)(og + 128 + i0) = ol;
    }
    __syncthreads();
  }
}

extern "C" void kernel_launch(void* const* d_in, const int* in_sizes, int n_in,
                              void* d_out, int out_size, void* d_ws, size_t ws_size,
                              hipStream_t stream) {
  const float* x    = (const float*)d_in[0];
  const float* B11  = (const float*)d_in[1];
  const float* B12  = (const float*)d_in[2];
  const float* B21  = (const float*)d_in[3];
  const float* B22  = (const float*)d_in[4];
  const float* logu = (const float*)d_in[5];
  const float* logl = (const float*)d_in[6];
  const float* Wk   = (const float*)d_in[7];
  const float* lb   = (const float*)d_in[8];
  const float* dk   = (const float*)d_in[9];
  const float* db   = (const float*)d_in[10];
  const int*   band = (const int*)d_in[11];

  bf16* Wsw  = (bf16*)d_ws;                            // 65536 bf16 = 128 KB
  bf16* Bbsw = (bf16*)((char*)d_ws + 65536 * 2);       // 131072 bf16 = 256 KB

  prep_w <<<256, 256, 0, stream>>>(Wk, Wsw);
  prep_bb<<<512, 256, 0, stream>>>(B11, B12, B21, B22, band, Bbsw);
  rkn_scan<<<NBLK, NTHREADS, LDS_TOTAL, stream>>>(x, logu, logl, lb, dk, db,
                                                  Wsw, Bbsw, (float*)d_out);
}
